// FeatLinear_85925115724421
// MI455X (gfx1250) — matrix operand order repaired and checked by reading, not compiled, not run on hardware
//
#include <hip/hip_runtime.h>
#include <hip/hip_bf16.h>
#include <stdint.h>

typedef __attribute__((ext_vector_type(16))) __bf16 v16bf;
typedef __attribute__((ext_vector_type(8)))  __bf16 v8bf;
typedef __attribute__((ext_vector_type(2)))  __bf16 v2bf;
typedef __attribute__((ext_vector_type(8)))  float  v8f;
typedef __attribute__((ext_vector_type(4)))  float  v4f;
typedef __attribute__((ext_vector_type(4)))  unsigned int v4u;

#define NB   4           // number of branches
#define OUTF 1024        // in/out feature width
#define ROWS (16 * 4096) // B*T
#define BM   32          // rows per block tile (each wave covers both 16-row halves)
#define BN   512         // outputs per block tile (8 waves x 64, no duplicate B loads)
#define KC   64          // K chunk staged in LDS (2 WMMA k-steps)
#define APAD 4           // fp32 pad -> 272B row stride, conflict-free 16B ds reads

// -------- workspace layout --------
// [0,16)               : per-branch row counters (4 x int)
// [256, 256+4*ROWS*4)  : per-branch row index bins
// [WOFF, WOFF+8MB)     : bf16-converted weights
#define BINS_OFF 256
#define WOFF (((BINS_OFF + NB * ROWS * 4) + 511) & ~511)

// ---------------- binning: row -> branch bucket ----------------
__global__ void bin_rows(const int* __restrict__ feat,
                         int* __restrict__ counts,
                         int* __restrict__ bins) {
    int r = blockIdx.x * blockDim.x + threadIdx.x;
    if (r >= ROWS) return;
    int v = feat[r];                 // 32 / 64 / 128 / 256
    int b = __builtin_ctz(v) - 5;    // -> 0..3
    b = (b < 0) ? 0 : (b > 3 ? 3 : b);
    int pos = atomicAdd(&counts[b], 1);
    bins[b * ROWS + pos] = r;
}

// pack two fp32 into 2x bf16 (RNE) using the hardware pack-cvt when available
__device__ __forceinline__ v2bf pk_bf16(float a, float b) {
#if __has_builtin(__builtin_amdgcn_cvt_pk_bf16_f32)
    return __builtin_bit_cast(v2bf, __builtin_amdgcn_cvt_pk_bf16_f32(a, b));
#else
    v2bf r; r[0] = (__bf16)a; r[1] = (__bf16)b; return r;
#endif
}

// ---------------- weights fp32 -> bf16 ----------------
__global__ void conv_w(const float* __restrict__ w, __bf16* __restrict__ wbf) {
    int i = (blockIdx.x * blockDim.x + threadIdx.x) * 8;
    v4f f0 = *(const v4f*)(w + i);
    v4f f1 = *(const v4f*)(w + i + 4);
    union { v8bf v; v2bf p[4]; } h;
    h.p[0] = pk_bf16(f0[0], f0[1]);
    h.p[1] = pk_bf16(f0[2], f0[3]);
    h.p[2] = pk_bf16(f1[0], f1[1]);
    h.p[3] = pk_bf16(f1[2], f1[3]);
    *(v8bf*)(wbf + i) = h.v;
}

// ---------------- gathered per-branch GEMM ----------------
__global__ __launch_bounds__(256) void gemm_branch(
    const float* __restrict__ x, const __bf16* __restrict__ wbf,
    const float* __restrict__ bias, const int* __restrict__ counts,
    const int* __restrict__ bins, float* __restrict__ out)
{
    const int branch = blockIdx.z;
    const int cnt = counts[branch];
    const int row0 = blockIdx.x * BM;
    if (row0 >= cnt) return;   // block-uniform early exit

    __shared__ __align__(16) float Axf[2][BM][KC + APAD]; // fp32 x tile, 272B stride
    __shared__ int rows[BM];

    const int tid  = threadIdx.x;
    const int lane = tid & 31;
    const int wn   = tid >> 5;    // 0..7 : unique N slice per wave

    if (tid < BM) {
        int g = row0 + tid;
        rows[tid] = (g < cnt) ? bins[branch * ROWS + g] : -1;
    }
    __syncthreads();

    // async staging: 32 rows x 64 fp32 cols; each thread copies 32B (2x b128)
    const int lrow = tid >> 3;
    const int lcol = (tid & 7) * 8;     // in floats
    const int grow = rows[lrow];
    auto issue_tile = [&](int kc, int buf) {
        float* dst = &Axf[buf][lrow][lcol];
        if (grow >= 0) {
            const float* src = x + (size_t)grow * OUTF + kc + lcol;
            unsigned lds = (unsigned)(uintptr_t)dst;
            // raw global->LDS copy, tracked by ASYNCcnt, bypasses VGPRs
            asm volatile(
                "global_load_async_to_lds_b128 %0, %1, off\n\t"
                "global_load_async_to_lds_b128 %0, %1, off offset:16"
                :: "v"(lds), "v"(src) : "memory");
        } else {
            v4u z = {0u, 0u, 0u, 0u};
            *(v4u*)dst = z;
            *(v4u*)(dst + 4) = z;
        }
    };

    v8f zero8 = {0.f, 0.f, 0.f, 0.f, 0.f, 0.f, 0.f, 0.f};
    v8f acc[2][4];
#pragma unroll
    for (int m = 0; m < 2; ++m)
#pragma unroll
        for (int i = 0; i < 4; ++i) acc[m][i] = zero8;

    issue_tile(0, 0);

    // A-fragment addressing (16-bit A 16x32): lane<16 -> K 0..15,
    // lane>=16 -> K 16..31 (matches B); row = m*16 + lane%16.
    const int arow  = lane & 15;
    const int koffA = (lane & 16) ? 16 : 0;
    const int koffB = (lane & 16) ? 16 : 0; // B: lane-half selects K 0..15 / 16..31
    const int nLane = lane & 15;
    const int nBase = blockIdx.y * BN + wn * 64;
    const __bf16* wrow = wbf + (size_t)(branch * OUTF) * OUTF;

    const int KCH = OUTF / KC; // 16 chunks
    for (int i = 0; i < KCH; ++i) {
        asm volatile("s_wait_asynccnt 0x0" ::: "memory"); // my buf-i copies landed
        __syncthreads();                                  // everyone's landed
        if (i + 1 < KCH) {
            issue_tile((i + 1) * KC, (i + 1) & 1);
            __builtin_prefetch(wrow + (size_t)(nBase + nLane) * OUTF + (i + 1) * KC, 0, 1);
        }
        const int buf = i & 1;
#pragma unroll
        for (int ks = 0; ks < KC; ks += 32) {
            // issue all 4 B-fragment loads first so they clause together
            const int kb = i * KC + ks + koffB;
            v16bf bfr[4];
#pragma unroll
            for (int nt = 0; nt < 4; ++nt) {
                const int n = nBase + nt * 16 + nLane;
                bfr[nt] = *(const v16bf*)(wrow + (size_t)n * OUTF + kb);
            }
            // build bf16 A fragments for both M halves with paired pack-cvt
            v16bf af[2];
#pragma unroll
            for (int m = 0; m < 2; ++m) {
                const float* ap = &Axf[buf][m * 16 + arow][ks + koffA];
                v4f f0 = *(const v4f*)(ap);
                v4f f1 = *(const v4f*)(ap + 4);
                v4f f2 = *(const v4f*)(ap + 8);
                v4f f3 = *(const v4f*)(ap + 12);
                union { v16bf v; v2bf p[8]; } u;
                u.p[0] = pk_bf16(f0[0], f0[1]);
                u.p[1] = pk_bf16(f0[2], f0[3]);
                u.p[2] = pk_bf16(f1[0], f1[1]);
                u.p[3] = pk_bf16(f1[2], f1[3]);
                u.p[4] = pk_bf16(f2[0], f2[1]);
                u.p[5] = pk_bf16(f2[2], f2[3]);
                u.p[6] = pk_bf16(f3[0], f3[1]);
                u.p[7] = pk_bf16(f3[2], f3[3]);
                af[m] = u.v;
            }
#pragma unroll
            for (int nt = 0; nt < 4; ++nt) {
#pragma unroll
                for (int m = 0; m < 2; ++m) {
                    acc[m][nt] = __builtin_amdgcn_wmma_f32_16x16x32_bf16(
                        false, af[m], false, bfr[nt], (short)0, acc[m][nt], false, false);
                }
            }
        }
    }

    // epilogue: + bias, NT scatter (out is write-once; don't pollute L2)
#pragma unroll
    for (int nt = 0; nt < 4; ++nt) {
        const int n = nBase + nt * 16 + nLane;
        const float bv = bias[branch * OUTF + n];
#pragma unroll
        for (int m = 0; m < 2; ++m) {
#pragma unroll
            for (int e = 0; e < 8; ++e) {
                const int rl = m * 16 + e + ((lane & 16) ? 8 : 0);
                const int gr = rows[rl];
                if (gr >= 0)
                    __builtin_nontemporal_store(acc[m][nt][e] + bv,
                                                &out[(size_t)gr * OUTF + n]);
            }
        }
    }
}

extern "C" void kernel_launch(void* const* d_in, const int* in_sizes, int n_in,
                              void* d_out, int out_size, void* d_ws, size_t ws_size,
                              hipStream_t stream) {
    const float* x    = (const float*)d_in[0];
    const int*   feat = (const int*)d_in[1];
    const float* w    = (const float*)d_in[2];
    const float* bias = (const float*)d_in[3];
    float* out = (float*)d_out;

    char* ws = (char*)d_ws;
    int*    counts = (int*)ws;
    int*    bins   = (int*)(ws + BINS_OFF);
    __bf16* wbf    = (__bf16*)(ws + WOFF);

    hipMemsetAsync(counts, 0, 16, stream);
    bin_rows<<<ROWS / 256, 256, 0, stream>>>(feat, counts, bins);
    conv_w<<<(NB * OUTF * OUTF) / (256 * 8), 256, 0, stream>>>(w, wbf);

    dim3 grid(ROWS / BM, OUTF / BN, NB);
    gemm_branch<<<grid, 256, 0, stream>>>(x, wbf, bias, counts, bins, out);
}
